// Attention_80693845557971
// MI455X (gfx1250) — matrix operand order repaired and checked by reading, not compiled, not run on hardware
//
#include <hip/hip_runtime.h>

// ---------------- problem constants ----------------
#define Bq   4
#define Sq   1024
#define Hq   32
#define KVHq 8
#define HDq  128
#define DIMq 4096

typedef unsigned int u32;
typedef unsigned short u16;
typedef __attribute__((ext_vector_type(16))) __bf16 bf16x16;
typedef __attribute__((ext_vector_type(8)))  float  f32x8;
typedef __attribute__((ext_vector_type(4)))  u32    u32x4;

// ---------------- helpers ----------------
static __device__ __forceinline__ u16 f2bf(float f) {
  u32 u = __float_as_uint(f);
  u += 0x7FFFu + ((u >> 16) & 1u);   // round-to-nearest-even
  return (u16)(u >> 16);
}

static __device__ __forceinline__ bf16x16 frag2(u32x4 lo, u32x4 hi) {
  union { u32x4 q[2]; bf16x16 v; } u;
  u.q[0] = lo; u.q[1] = hi;
  return u.v;
}

// A-matrix 16x32 bf16 fragment: lane holds row (lane&15);
// contiguous K chunk of 16 at 16*(lane>>4).
static __device__ __forceinline__ bf16x16 loadFragA(const u16* p) {
  return frag2(*(const u32x4*)p, *(const u32x4*)(p + 8));
}

// B-matrix 32x16 bf16 fragment: lane holds col (lane&15);
// contiguous K chunk of 16 at 16*(lane>>4).
static __device__ __forceinline__ bf16x16 loadFragB(const u16* p) {
  return frag2(*(const u32x4*)p, *(const u32x4*)(p + 8));
}

static __device__ __forceinline__ f32x8 wmma_bf(bf16x16 a, bf16x16 b, f32x8 c) {
  // (neg_a, A, neg_b, B, c_mod, C, reuse_a, reuse_b)
  return __builtin_amdgcn_wmma_f32_16x16x32_bf16(false, a, false, b, (short)0, c,
                                                 false, false);
}

// Async global -> LDS copy of 16 bytes (CDNA5 GLOBAL_LOAD_ASYNC_TO_LDS_B128).
// VDST = VGPR with wave-relative LDS byte offset; VADDR = 64-bit global address.
// Generic LDS pointers carry the LDS offset in their low 32 bits (flat->LDS
// truncates the upper dword), so (u32)(size_t)p is the VDST payload.
static __device__ __forceinline__ void async_b128(u16* lds_p, const u16* gp) {
  u32 ldsoff = (u32)(size_t)lds_p;
  asm volatile("global_load_async_to_lds_b128 %0, %1, off"
               :: "v"(ldsoff), "v"(gp)
               : "memory");
}
static __device__ __forceinline__ void wait_async0() {
  asm volatile("s_wait_asynccnt 0x0" ::: "memory");
}

// ---------------- elementwise kernels ----------------
__global__ __launch_bounds__(256) void cvt_f32_bf16(const float* __restrict__ in,
                                                    u16* __restrict__ out, int n) {
  int i = blockIdx.x * 256 + threadIdx.x;
  if (i < n) out[i] = f2bf(in[i]);
}

// in: f32 [K, N] row-major -> out: bf16 [N, K] row-major (transpose + convert)
__global__ __launch_bounds__(256) void tconv_f32_bf16(const float* __restrict__ in,
                                                      u16* __restrict__ out,
                                                      int K, int N) {
  int idx = blockIdx.x * 256 + threadIdx.x;
  if (idx >= K * N) return;
  int n = idx % N;
  int k = idx / N;
  out[(size_t)n * K + k] = f2bf(in[idx]);
}

// RoPE + scale + bf16 cast. Layout [B,S,nh,128]; one thread per (token,head,pair).
__global__ __launch_bounds__(256) void rope_bf16(const float* __restrict__ in,
                                                 u16* __restrict__ out,
                                                 int nheads, float scale) {
  int idx = blockIdx.x * 256 + threadIdx.x;   // idx = ((b*S+s)*nh + h)*64 + j
  int j = idx & 63;
  int rest = idx >> 6;                        // (b*S+s)*nh + h
  int s = (rest / nheads) & (Sq - 1);
  float freq = __expf(-(float)(2 * j) * (9.210340371976184f / 128.0f)); // theta^(-2j/128)
  float ang = (float)s * freq;
  float sn, cs;
  __sincosf(ang, &sn, &cs);
  float xr = in[2 * (size_t)idx];
  float xi = in[2 * (size_t)idx + 1];
  out[2 * (size_t)idx]     = f2bf((xr * cs - xi * sn) * scale);
  out[2 * (size_t)idx + 1] = f2bf((xr * sn + xi * cs) * scale);
}

// ---------------- WMMA GEMM: C[M,N](f32) = A[M,K](bf16) * Bt[N,K](bf16)^T ----
// Block = 256 threads (8 waves), tile 128(M) x 64(N); wave tile 32x32 = 2x2 WMMA.
__global__ __launch_bounds__(256) void gemm_bf16(const u16* __restrict__ A,
                                                 const u16* __restrict__ Bt,
                                                 float* __restrict__ C,
                                                 int M, int N, int K) {
  const int lane = threadIdx.x & 31;
  const int wv   = threadIdx.x >> 5;
  const int ln   = lane & 15;
  const int hg   = lane >> 4;
  const int m0 = blockIdx.y * 128 + (wv >> 1) * 32;
  const int n0 = blockIdx.x * 64  + (wv & 1)  * 32;

  f32x8 c00 = {}, c01 = {}, c10 = {}, c11 = {};

  const u16* a0 = A  + (size_t)(m0 + ln)      * K + hg * 16;
  const u16* a1 = A  + (size_t)(m0 + 16 + ln) * K + hg * 16;
  const u16* b0 = Bt + (size_t)(n0 + ln)      * K + hg * 16;
  const u16* b1 = Bt + (size_t)(n0 + 16 + ln) * K + hg * 16;

  for (int k0 = 0; k0 < K; k0 += 32) {
    // pull rows ~8 k-steps ahead into cache (speculative; OOB dropped by HW)
    __builtin_prefetch(a0 + k0 + 256, 0, 1);
    __builtin_prefetch(a1 + k0 + 256, 0, 1);
    __builtin_prefetch(b0 + k0 + 256, 0, 1);
    __builtin_prefetch(b1 + k0 + 256, 0, 1);

    bf16x16 af0 = loadFragA(a0 + k0);
    bf16x16 af1 = loadFragA(a1 + k0);
    bf16x16 bf0 = loadFragB(b0 + k0);
    bf16x16 bf1 = loadFragB(b1 + k0);
    c00 = wmma_bf(af0, bf0, c00);
    c01 = wmma_bf(af0, bf1, c01);
    c10 = wmma_bf(af1, bf0, c10);
    c11 = wmma_bf(af1, bf1, c11);
  }

#pragma unroll
  for (int i = 0; i < 8; ++i) {
    int r0 = m0 + i + 8 * hg;
    int r1 = r0 + 16;
    C[(size_t)r0 * N + n0 + ln]      = c00[i];
    C[(size_t)r0 * N + n0 + 16 + ln] = c01[i];
    C[(size_t)r1 * N + n0 + ln]      = c10[i];
    C[(size_t)r1 * N + n0 + 16 + ln] = c11[i];
  }
}

// ---------------- Flash attention (causal, GQA), WMMA bf16 ----------------
// Grid: (S/128, H, B). Block 256 = 8 waves; wave w owns q rows [q0+16w, q0+16w+16).
// Q pre-scaled by 1/sqrt(HD). Output written directly as bf16 [B*S, H*HD].
// K tile staged via async global->LDS DMA (all 8 waves share one KV head);
// V tile staged transposed so PV B-fragments are contiguous LDS reads.
__global__ __launch_bounds__(256) void attn_bf16(const u16* __restrict__ Q,
                                                 const u16* __restrict__ Kc,
                                                 const u16* __restrict__ Vc,
                                                 u16* __restrict__ Ob) {
  __shared__ __attribute__((aligned(16))) u16 Kt[32 * HDq];   // K tile [key][d]
  __shared__ __attribute__((aligned(16))) u16 Vt[HDq * 32];   // V^T tile [d][key]
  __shared__ __attribute__((aligned(16))) u16 Pws[8][16 * 32];// per-wave P scratch

  const int tid  = threadIdx.x;
  const int lane = tid & 31;
  const int wv   = tid >> 5;
  const int ln   = lane & 15;
  const int hg   = lane >> 4;
  const int b    = blockIdx.z;
  const int h    = blockIdx.y;
  const int kvh  = h >> 2;            // N_REP = 4
  const int qr0  = blockIdx.x * 128 + wv * 16;

  // Q fragments: head dim 128 = 4 chunks of K=32
  bf16x16 qf[4];
  {
    const u16* qbase = Q + (((size_t)(b * Sq + qr0 + ln)) * Hq + h) * HDq + hg * 16;
#pragma unroll
    for (int kc = 0; kc < 4; ++kc) qf[kc] = loadFragA(qbase + kc * 32);
  }

  float mi[8], li[8];
  f32x8 Oacc[8];
  {
    f32x8 z = {};
#pragma unroll
    for (int i = 0; i < 8; ++i) { mi[i] = -3.0e38f; li[i] = 0.0f; Oacc[i] = z; }
  }

  const int kend = blockIdx.x * 128 + 128;  // causal: keys < kend
  for (int kb = 0; kb < kend; kb += 32) {
    __syncthreads();

    // K tile: async DMA, 512 x 16B chunks; 2 per thread.
    // chunk idx -> key = idx/16, c = idx%16 (16 b128 chunks per 256B row)
#pragma unroll
    for (int t = 0; t < 2; ++t) {
      int idx = tid + t * 256;
      int key = idx >> 4, c = idx & 15;
      const u16* gp =
          Kc + (((size_t)(b * Sq + kb + key)) * KVHq + kvh) * HDq + c * 8;
      async_b128(&Kt[key * HDq + c * 8], gp);
    }

    // V tile: cooperative transpose into LDS (coalesced read over d)
    for (int idx = tid; idx < 32 * HDq; idx += 256) {
      int key = idx >> 7, d = idx & 127;
      Vt[d * 32 + key] =
          Vc[(((size_t)(b * Sq + kb + key)) * KVHq + kvh) * HDq + d];
    }

    wait_async0();
    __syncthreads();

    // scores: S = Q * K^T for 2 n-tiles of 16 keys (K fragments from LDS)
    f32x8 s0 = {}, s1 = {};
#pragma unroll
    for (int kc = 0; kc < 4; ++kc) {
      const u16* k0p = &Kt[(ln)      * HDq + kc * 32 + hg * 16];
      const u16* k1p = &Kt[(16 + ln) * HDq + kc * 32 + hg * 16];
      s0 = wmma_bf(qf[kc], loadFragB(k0p), s0);
      s1 = wmma_bf(qf[kc], loadFragB(k1p), s1);
    }

    // online softmax; C layout: vgpr i holds row (i + 8*hg), col ln
    u16* pw = &Pws[wv][0];
#pragma unroll
    for (int i = 0; i < 8; ++i) {
      int row = qr0 + i + 8 * hg;
      float x0 = (kb + ln      <= row) ? s0[i] : -1.0e30f;
      float x1 = (kb + 16 + ln <= row) ? s1[i] : -1.0e30f;
      float t = fmaxf(x0, x1);
      t = fmaxf(t, __shfl_xor(t, 1, 16));
      t = fmaxf(t, __shfl_xor(t, 2, 16));
      t = fmaxf(t, __shfl_xor(t, 4, 16));
      t = fmaxf(t, __shfl_xor(t, 8, 16));
      float mnew  = fmaxf(mi[i], t);
      float alpha = __expf(mi[i] - mnew);
      float p0 = __expf(x0 - mnew);
      float p1 = __expf(x1 - mnew);
      float rs = p0 + p1;
      rs += __shfl_xor(rs, 1, 16);
      rs += __shfl_xor(rs, 2, 16);
      rs += __shfl_xor(rs, 4, 16);
      rs += __shfl_xor(rs, 8, 16);
      li[i] = li[i] * alpha + rs;
      mi[i] = mnew;
#pragma unroll
      for (int t8 = 0; t8 < 8; ++t8) Oacc[t8][i] *= alpha;
      pw[(i + 8 * hg) * 32 + ln]      = f2bf(p0);
      pw[(i + 8 * hg) * 32 + 16 + ln] = f2bf(p1);
    }

    // P as A-fragment (16x32), V^T chunks as B-fragments from LDS
    bf16x16 pf = loadFragA(pw + ln * 32 + hg * 16);
#pragma unroll
    for (int t8 = 0; t8 < 8; ++t8) {
      const u16* vp = &Vt[(t8 * 16 + ln) * 32 + hg * 16];
      Oacc[t8] = wmma_bf(pf, loadFragB(vp), Oacc[t8]);
    }
  }

  // epilogue: normalize and store bf16 [B*S, H*HD]
#pragma unroll
  for (int i = 0; i < 8; ++i) {
    int srow = qr0 + i + 8 * hg;
    float inv = 1.0f / li[i];
    u16* op = Ob + ((size_t)(b * Sq + srow)) * DIMq + h * HDq;
#pragma unroll
    for (int t8 = 0; t8 < 8; ++t8) op[t8 * 16 + ln] = f2bf(Oacc[t8][i] * inv);
  }
}

// ---------------- launcher ----------------
extern "C" void kernel_launch(void* const* d_in, const int* in_sizes, int n_in,
                              void* d_out, int out_size, void* d_ws, size_t ws_size,
                              hipStream_t stream) {
  (void)in_sizes; (void)n_in; (void)out_size; (void)ws_size;
  const float* x  = (const float*)d_in[0];
  const float* wq = (const float*)d_in[1];
  const float* wk = (const float*)d_in[2];
  const float* wv = (const float*)d_in[3];
  const float* wo = (const float*)d_in[4];
  float* out = (float*)d_out;

  char* ws = (char*)d_ws;
  const size_t MB = (size_t)1 << 20;
  u16*   xb  = (u16*)(ws + 0);          //  32 MB  x bf16 [4096,4096]
  u16*   wqt = (u16*)(ws + 32 * MB);    //  32 MB  wq^T bf16 [4096,4096]
  u16*   wkt = (u16*)(ws + 64 * MB);    //   8 MB  wk^T bf16 [1024,4096]
  u16*   wvt = (u16*)(ws + 72 * MB);    //   8 MB  wv^T bf16 [1024,4096]
  u16*   wot = (u16*)(ws + 80 * MB);    //  32 MB  wo^T bf16 [4096,4096]
  float* qf  = (float*)(ws + 112 * MB); //  64 MB  q f32 [4096,4096]
  float* kf  = (float*)(ws + 176 * MB); //  16 MB  k f32 [4096,1024]
  float* vf  = (float*)(ws + 192 * MB); //  16 MB  v f32 [4096,1024]
  u16*   qb  = (u16*)(ws + 208 * MB);   //  32 MB  q bf16 (roped+scaled)
  u16*   kb  = (u16*)(ws + 240 * MB);   //   8 MB  k bf16 (roped)
  u16*   vb  = (u16*)(ws + 248 * MB);   //   8 MB  v bf16
  u16*   ab  = (u16*)(ws + 112 * MB);   //  32 MB  attn out bf16 (reuses qf region)

  const int nX  = DIMq * Bq * Sq;       // 16,777,216
  const int nKV = Bq * Sq * KVHq * HDq; //  4,194,304

  cvt_f32_bf16<<<nX / 256, 256, 0, stream>>>(x, xb, nX);
  tconv_f32_bf16<<<nX / 256, 256, 0, stream>>>(wq, wqt, DIMq, DIMq);
  tconv_f32_bf16<<<nKV / 256, 256, 0, stream>>>(wk, wkt, DIMq, KVHq * HDq);
  tconv_f32_bf16<<<nKV / 256, 256, 0, stream>>>(wv, wvt, DIMq, KVHq * HDq);
  tconv_f32_bf16<<<nX / 256, 256, 0, stream>>>(wo, wot, DIMq, DIMq);

  // projections: M=4096 tokens, K=4096
  gemm_bf16<<<dim3(DIMq / 64, (Bq * Sq) / 128), 256, 0, stream>>>(
      xb, wqt, qf, Bq * Sq, DIMq, DIMq);
  gemm_bf16<<<dim3((KVHq * HDq) / 64, (Bq * Sq) / 128), 256, 0, stream>>>(
      xb, wkt, kf, Bq * Sq, KVHq * HDq, DIMq);
  gemm_bf16<<<dim3((KVHq * HDq) / 64, (Bq * Sq) / 128), 256, 0, stream>>>(
      xb, wvt, vf, Bq * Sq, KVHq * HDq, DIMq);

  // RoPE (+ fold 1/sqrt(HD) into q) and bf16 casts
  rope_bf16<<<(Bq * Sq * Hq * 64) / 256, 256, 0, stream>>>(
      qf, qb, Hq, 0.08838834764831845f);
  rope_bf16<<<(Bq * Sq * KVHq * 64) / 256, 256, 0, stream>>>(kf, kb, KVHq, 1.0f);
  cvt_f32_bf16<<<nKV / 256, 256, 0, stream>>>(vf, vb, nKV);

  // causal flash attention
  attn_bf16<<<dim3(Sq / 128, Hq, Bq), 256, 0, stream>>>(qb, kb, vb, ab);

  // output projection -> f32 result
  gemm_bf16<<<dim3(DIMq / 64, (Bq * Sq) / 128), 256, 0, stream>>>(
      ab, wot, out, Bq * Sq, DIMq, DIMq);
}
